// MultiHeadGraphAttentionLayer_15135464751290
// MI455X (gfx1250) — matrix operand order repaired and checked by reading, not compiled, not run on hardware
//
#include <hip/hip_runtime.h>
#include <hip/hip_bf16.h>
#include <cstdint>

#define NEG_SLOPE 0.2f
#define LN_EPS 1e-5f

constexpr int B = 4, N = 2048, D = 256, H = 4, HD = 64;

typedef __attribute__((ext_vector_type(16))) __bf16 v16bf;
typedef __attribute__((ext_vector_type(8)))  __bf16 v8bf;
typedef __attribute__((ext_vector_type(8)))  float  v8f;

__device__ __forceinline__ float lane_get(float v, int src) { return __shfl(v, src, 32); }

// ---------------- prep: x -> bf16 ----------------
__global__ void k_cvt_x(const float* __restrict__ x, __bf16* __restrict__ xb, int n) {
    int i = blockIdx.x * 256 + threadIdx.x;
    if (i < n) xb[i] = (__bf16)x[i];
}

// ---------------- prep: W (H,D,hd) -> W^T bf16 [h][d][k] ----------------
__global__ void k_cvt_w(const float* __restrict__ W, __bf16* __restrict__ wT) {
    int t = blockIdx.x * 256 + threadIdx.x;       // H*HD*D threads
    int k = t % D;
    int d = (t / D) % HD;
    int h = t / (D * HD);
    wT[t] = (__bf16)W[((size_t)h * D + k) * HD + d];
}

// ---------------- Wh^T = (x @ W)^T, bf16, layout [b][h][d][j] ----------------
__global__ void __launch_bounds__(128) k_whT(const __bf16* __restrict__ xb,
                                             const __bf16* __restrict__ wT,
                                             __bf16* __restrict__ whT) {
    const int i0 = blockIdx.x * 16;      // token-row tile
    const int bh = blockIdx.y;           // b*H + h
    const int b = bh / H, h = bh % H;
    const int w = threadIdx.x >> 5;      // wave -> d-tile 0..3
    const int L = threadIdx.x & 31;
    const int n = L & 15, hi = L >> 4;

    const __bf16* xrow = xb + ((size_t)b * N + i0 + n) * D;       // A: row m = L%16
    const __bf16* wrow = wT + ((size_t)h * HD + w * 16 + n) * D;  // B: fixed d, K contiguous

    v8f acc = {};
#pragma unroll
    for (int k0 = 0; k0 < D; k0 += 32) {
        v16bf af = *(const v16bf*)(xrow + k0 + hi * 16);      // K = k0 + hi*16 + e
        v16bf bf = *(const v16bf*)(wrow + k0 + hi * 16);      // K = k0 + hi*16 + e
        acc = __builtin_amdgcn_wmma_f32_16x16x32_bf16(false, af, false, bf,
                                                      (short)0, acc, false, false);
    }

    // transpose through LDS -> coalesced bf16 store of Wh^T
    __shared__ __bf16 t[4][16][20];
#pragma unroll
    for (int r = 0; r < 8; ++r)
        t[w][n][r + 8 * hi] = (__bf16)acc[r];   // element (j_local = r+8*hi, d_local = n)
    __syncthreads();
    int dl = L >> 1, seg = L & 1;
    v8bf o;
#pragma unroll
    for (int e = 0; e < 8; ++e) o[e] = t[w][dl][seg * 8 + e];
    *(v8bf*)(whT + ((size_t)bh * HD + w * 16 + dl) * N + i0 + seg * 8) = o;
}

// ---------------- ei/ej = Wh . a1 / a2 (coalesced over j) ----------------
__global__ void k_eiej(const __bf16* __restrict__ whT, const float* __restrict__ a,
                       float* __restrict__ ei, float* __restrict__ ej) {
    int j = blockIdx.x * 256 + threadIdx.x;
    int bh = blockIdx.y, h = bh % H;
    const __bf16* wp = whT + (size_t)bh * HD * N + j;
    const float* ah = a + (size_t)h * 2 * HD;
    float s1 = 0.f, s2 = 0.f;
    for (int d = 0; d < HD; ++d) {
        float v = (float)wp[(size_t)d * N];
        s1 += v * ah[d];
        s2 += v * ah[HD + d];
    }
    ei[(size_t)bh * N + j] = s1;
    ej[(size_t)bh * N + j] = s2;
}

// ---------------- fused flash-GAT + residual + LayerNorm ----------------
__global__ void __launch_bounds__(512) k_attn(const float* __restrict__ x,
                                              const int* __restrict__ adj,
                                              const __bf16* __restrict__ whT,
                                              const float* __restrict__ ei,
                                              const float* __restrict__ ej,
                                              const float* __restrict__ gamma,
                                              const float* __restrict__ beta,
                                              float* __restrict__ out) {
    const int i0 = blockIdx.x * 16;
    const int b = blockIdx.y;
    const int tid = threadIdx.x;
    const int wv = tid >> 5;       // 16 waves
    const int h = wv >> 2;         // head
    const int c = wv & 3;          // d-tile
    const int L = tid & 31;
    const int n = L & 15, hi = L >> 4;
    const int bh = b * H + h;

    __shared__ int adjs[2][16][32];
    __shared__ float oy[16][D];

    const int* adjb = adj + (size_t)b * N * N;
    const float eiv = ei[(size_t)bh * N + i0 + n];
    const float* ejrow = ej + (size_t)bh * N;
    const __bf16* brow = whT + ((size_t)bh * HD + c * 16 + n) * N;

    auto prefetch = [&](int jt, int buf) {
        if (tid < 128) {                              // waves 0..3, wave-uniform
            int row = tid >> 3, seg = tid & 7;
            uint32_t lds = (uint32_t)(uintptr_t)&adjs[buf][row][seg * 4];
            const int* gp = adjb + (size_t)(i0 + row) * N + jt * 32 + seg * 4;
            asm volatile("global_load_async_to_lds_b128 %0, %1, off"
                         :: "v"(lds), "v"(gp) : "memory");
        }
    };

    v8f acc = {};
    float mrun = -INFINITY, lrun = 0.f;
    const int NT = N / 32;

    prefetch(0, 0);
    for (int jt = 0; jt < NT; ++jt) {
        const int buf = jt & 1;
        if (jt + 1 < NT) {
            prefetch(jt + 1, buf ^ 1);                     // overlap next tile
            asm volatile("s_wait_asynccnt 0x1" ::: "memory");  // wait: current tile landed
        } else {
            asm volatile("s_wait_asynccnt 0x0" ::: "memory");
        }
        __syncthreads();                                   // tile jt visible to all waves

        const int j0 = jt * 32;
        float p[16];
        float mloc = -INFINITY;
#pragma unroll
        for (int rp = 0; rp < 2; ++rp) {                   // A-fragment K runs
            const int jb = hi * 16 + rp * 8;
            const float4* ejp = (const float4*)(ejrow + j0 + jb);
            const int4* ap = (const int4*)&adjs[buf][n][jb];
#pragma unroll
            for (int q = 0; q < 2; ++q) {
                float4 ev = ejp[q];
                int4 av = ap[q];
                float ss[4] = {eiv + ev.x, eiv + ev.y, eiv + ev.z, eiv + ev.w};
                int aa[4] = {av.x, av.y, av.z, av.w};
#pragma unroll
                for (int u = 0; u < 4; ++u) {
                    float s = ss[u];
                    s = (s >= 0.f) ? s : NEG_SLOPE * s;    // leaky relu
                    s = aa[u] ? s : -INFINITY;             // adjacency mask
                    p[rp * 8 + q * 4 + u] = s;
                    mloc = fmaxf(mloc, s);
                }
            }
        }
        // online softmax: row stats live in lane pair (m, m+16)
        mloc = fmaxf(mloc, __shfl_xor(mloc, 16, 32));
        float mnew = fmaxf(mrun, mloc);
        float scale = (mnew == -INFINITY) ? 1.f : __expf(mrun - mnew);
        float sloc = 0.f;
        v16bf af;
#pragma unroll
        for (int e = 0; e < 16; ++e) {
            float pe = (p[e] == -INFINITY) ? 0.f : __expf(p[e] - mnew);
            sloc += pe;
            af[e] = (__bf16)pe;
        }
        sloc += __shfl_xor(sloc, 16, 32);
        lrun = lrun * scale + sloc;
        mrun = mnew;

        if (__any(scale != 1.f)) {                         // rescale O accumulator
#pragma unroll
            for (int r = 0; r < 8; ++r)
                acc[r] *= lane_get(scale, r + 8 * hi);
        }

        v16bf bf = *(const v16bf*)(brow + j0 + hi * 16);   // B: Wh^T, K contiguous
        acc = __builtin_amdgcn_wmma_f32_16x16x32_bf16(false, af, false, bf,
                                                      (short)0, acc, false, false);
        __syncthreads();                                   // adjs[buf] free for reuse
    }

    // normalize by l and stage O tile in LDS
#pragma unroll
    for (int r = 0; r < 8; ++r) {
        float lr = lane_get(lrun, r + 8 * hi);
        oy[r + 8 * hi][h * HD + c * 16 + n] = (lr > 0.f) ? acc[r] / lr : 0.f;
    }
    __syncthreads();

    // fused residual + LayerNorm: wave wv handles row wv (D=256, 8 vals/lane)
    {
        const int row = wv;
        const float* xr = x + ((size_t)b * N + i0 + row) * D;
        float y[8];
        float sum = 0.f, sq = 0.f;
#pragma unroll
        for (int e = 0; e < 8; ++e) {
            float v = xr[L * 8 + e] + oy[row][L * 8 + e];
            y[e] = v;
            sum += v;
            sq += v * v;
        }
#pragma unroll
        for (int off = 16; off >= 1; off >>= 1) {
            sum += __shfl_xor(sum, off, 32);
            sq += __shfl_xor(sq, off, 32);
        }
        float mean = sum * (1.f / D);
        float var = sq * (1.f / D) - mean * mean;
        float inv = rsqrtf(var + LN_EPS);
        float* orow = out + ((size_t)b * N + i0 + row) * D;
#pragma unroll
        for (int e = 0; e < 8; ++e) {
            int d = L * 8 + e;
            orow[d] = (y[e] - mean) * inv * gamma[d] + beta[d];
        }
    }
}

extern "C" void kernel_launch(void* const* d_in, const int* in_sizes, int n_in,
                              void* d_out, int out_size, void* d_ws, size_t ws_size,
                              hipStream_t stream) {
    const float* x     = (const float*)d_in[0];
    const int*   adj   = (const int*)d_in[1];
    const float* W     = (const float*)d_in[2];
    const float* a     = (const float*)d_in[3];
    const float* gamma = (const float*)d_in[4];
    const float* beta  = (const float*)d_in[5];
    float* out = (float*)d_out;

    char* ws = (char*)d_ws;
    __bf16* xb  = (__bf16*)ws; ws += (size_t)B * N * D * 2;       // 4 MB
    __bf16* wT  = (__bf16*)ws; ws += (size_t)H * HD * D * 2;      // 128 KB
    __bf16* whT = (__bf16*)ws; ws += (size_t)B * H * HD * N * 2;  // 4 MB
    float*  ei  = (float*)ws;  ws += (size_t)B * H * N * 4;       // 128 KB
    float*  ej  = (float*)ws;                                     // 128 KB

    k_cvt_x<<<(B * N * D) / 256, 256, 0, stream>>>(x, xb, B * N * D);
    k_cvt_w<<<(H * HD * D) / 256, 256, 0, stream>>>(W, wT);
    k_whT<<<dim3(N / 16, B * H), 128, 0, stream>>>(xb, wT, whT);
    k_eiej<<<dim3(N / 256, B * H), 256, 0, stream>>>(whT, a, ei, ej);
    k_attn<<<dim3(N / 16, B), 512, 0, stream>>>(x, adj, whT, ei, ej, gamma, beta, out);
}
